// Block_75986561401374
// MI455X (gfx1250) — matrix operand order repaired and checked by reading, not compiled, not run on hardware
//
#include <hip/hip_runtime.h>
#include <math.h>

// ---------------- problem constants ----------------
#define NN 2048          // nodes
#define BB 16            // batch
#define LL 12            // timesteps
#define CC 32            // in channels
#define DD 32            // out channels
#define MM (BB*LL*CC)    // 6144 : columns of propagation GEMM  (j = (b,t,c), c fastest)
#define RR (BB*NN*LL)    // 393216 : rows of row-major GEMMs (r = (b,n,t), t fastest)

typedef __bf16 bf16;
typedef __attribute__((ext_vector_type(16))) __bf16 v16bf;
typedef __attribute__((ext_vector_type(8)))  __bf16 v8bf;
typedef __attribute__((ext_vector_type(8)))  float  v8f;

// ---------------- bf16 helpers (explicit RNE bit ops) ----------------
__device__ __forceinline__ bf16 f2bf(float f) {
    unsigned u = __float_as_uint(f);
    u += 0x7FFFu + ((u >> 16) & 1u);
    unsigned short h = (unsigned short)(u >> 16);
    return __builtin_bit_cast(bf16, h);
}
__device__ __forceinline__ float bf2f(bf16 h) {
    unsigned short s = __builtin_bit_cast(unsigned short, h);
    return __uint_as_float(((unsigned)s) << 16);
}
__device__ __forceinline__ bf16 bfzero() {
    return __builtin_bit_cast(bf16, (unsigned short)0);
}
__device__ __forceinline__ v16bf cat8(v8bf lo, v8bf hi) {
    v16bf r;
#pragma unroll
    for (int e = 0; e < 8; ++e) { r[e] = lo[e]; r[8 + e] = hi[e]; }
    return r;
}

// A-matrix (16x32 bf16, MxK) per-lane load.
// lane<16 : M=lane,    elements 0..7 -> K=0..7,   8..15 -> K=16..23
// lane>=16: M=lane-16, elements 0..7 -> K=8..15,  8..15 -> K=24..31
__device__ __forceinline__ v16bf load_a_bf(const bf16* base, int khalf) {
    v8bf lo = *(const v8bf*)(base + khalf * 8);
    v8bf hi = *(const v8bf*)(base + 16 + khalf * 8);
    return cat8(lo, hi);
}
// B-matrix (32x16 bf16, KxN): lane = N col (lane&15), elements follow A's K order (stride ld).
__device__ __forceinline__ v16bf load_b_bf(const bf16* q, size_t ld, int khalf) {
    v16bf r;
#pragma unroll
    for (int e = 0; e < 8; ++e) {
        r[e]     = q[(size_t)(khalf * 8 + e) * ld];
        r[8 + e] = q[(size_t)(16 + khalf * 8 + e) * ld];
    }
    return r;
}
#define WMMA_BF16(A, Bm, Cv) \
    __builtin_amdgcn_wmma_f32_16x16x32_bf16(false, (A), false, (Bm), (short)0, (Cv), false, false)

// ---------------- small prep kernels ----------------
__global__ void deg_kernel(const float* __restrict__ A, float* __restrict__ dinv) {
    int wid = (blockIdx.x * blockDim.x + threadIdx.x) >> 5;
    int lane = threadIdx.x & 31;
    if (wid >= NN) return;
    const float* row = A + (size_t)wid * NN;
    float s = 0.f;
    for (int m = lane; m < NN; m += 32) s += row[m];
    for (int off = 16; off; off >>= 1) s += __shfl_down(s, off, 32);
    if (lane == 0) dinv[wid] = rsqrtf(s);
}

__global__ void build_lap(const float* __restrict__ A, const float* __restrict__ dinv,
                          float* __restrict__ Lap) {
    int i = blockIdx.x * blockDim.x + threadIdx.x;   // NN*NN threads exactly
    int n = i >> 11, m = i & (NN - 1);
    Lap[i] = (n == m ? 1.f : 0.f) - dinv[n] * A[i] * dinv[m];
}

__global__ void init_v(float* __restrict__ v) {
    int i = blockIdx.x * blockDim.x + threadIdx.x;
    if (i < NN) v[i] = (i & 1) ? -1.f : 1.f;   // oscillatory start -> overlaps top eigvec
}

__global__ void matvec(const float* __restrict__ Lap, const float* __restrict__ v,
                       float* __restrict__ y) {
    int wid = (blockIdx.x * blockDim.x + threadIdx.x) >> 5;
    int lane = threadIdx.x & 31;
    if (wid >= NN) return;
    const float* row = Lap + (size_t)wid * NN;
    float s = 0.f;
    for (int m = lane; m < NN; m += 32) s += row[m] * v[m];
    for (int off = 16; off; off >>= 1) s += __shfl_down(s, off, 32);
    if (lane == 0) y[wid] = s;
}

__global__ void norm_scale(const float* __restrict__ y, float* __restrict__ v,
                           float* __restrict__ lmax) {
    __shared__ float red[256];
    int tid = threadIdx.x;
    float s = 0.f;
    for (int i = tid; i < NN; i += 256) s += y[i] * y[i];
    red[tid] = s;
    __syncthreads();
    for (int st = 128; st > 0; st >>= 1) {
        if (tid < st) red[tid] += red[tid + st];
        __syncthreads();
    }
    float nrm = sqrtf(red[0]);
    if (tid == 0) *lmax = nrm;                  // ||Lap v|| -> lambda_max
    float inv = 1.f / (nrm + 1e-20f);
    for (int i = tid; i < NN; i += 256) v[i] = y[i] * inv;
}

__global__ void build_ls(const float* __restrict__ Lap, const float* __restrict__ lmax,
                         bf16* __restrict__ Ls) {
    int i = blockIdx.x * blockDim.x + threadIdx.x;   // NN*NN threads
    int n = i >> 11, m = i & (NN - 1);
    float sc = 2.f / lmax[0];
    Ls[i] = f2bf(sc * Lap[i] - (n == m ? 1.f : 0.f));
}

__global__ void cvt_copy_bf(const float* __restrict__ src, bf16* __restrict__ dst, int n) {
    int i = blockIdx.x * blockDim.x + threadIdx.x;
    if (i < n) dst[i] = f2bf(src[i]);
}

// x[B,N,L,C] -> X_bf [N][M] with j=(b,t,c), c fastest (coalesced stores)
__global__ void pack_x_bf(const float* __restrict__ x, bf16* __restrict__ X) {
    int i = blockIdx.x * blockDim.x + threadIdx.x;   // NN*MM threads exactly
    int n = i / MM;
    int j = i - n * MM;
    int bt = j >> 5, c = j & 31;
    int b = bt / LL, t = bt - b * LL;
    X[i] = f2bf(x[(((size_t)b * NN + n) * LL + t) * CC + c]);
}

// tcn_w [conv][o][i][kn][kl] -> Wt[conv][(kn*2+kl)*32 + i][o]   (GEMM-B layout)
__global__ void cvt_tcn_w(const float* __restrict__ w, bf16* __restrict__ Wt) {
    int idx = blockIdx.x * blockDim.x + threadIdx.x;
    if (idx >= 4 * 32 * 32 * 4) return;
    int conv = idx >> 12;
    int rem  = idx & 4095;
    int o  = rem >> 7;
    int r2 = rem & 127;
    int i  = r2 >> 2;
    int kn = (r2 >> 1) & 1;
    int kl = r2 & 1;
    Wt[(size_t)conv * 4096 + ((kn * 2 + kl) * 32 + i) * 32 + o] = f2bf(w[idx]);
}

// -------- propagation GEMM: Out = alpha * Ls @ S + beta * T  (all bf16 operands) --------
// 32-row x 64-col tile per wave: 8 WMMA per K-step fed by 2 A-loads + 4 B-loads.
__global__ __launch_bounds__(256) void prop_gemm(
    const bf16* __restrict__ Ls, const bf16* __restrict__ S,
    const bf16* __restrict__ Tb, int t_mode,
    float alpha, float beta, bf16* __restrict__ Out)
{
    int wave = (blockIdx.x * blockDim.x + threadIdx.x) >> 5;
    int lane = threadIdx.x & 31;
    const int CGS = MM / 64;                   // 96 column groups
    int tileR = wave / CGS;
    int cg    = wave % CGS;
    if (tileR >= NN / 32) return;              // wave-uniform guard
    int n0 = tileR * 32;
    int j0 = cg * 64;
    int khalf = lane >> 4;
    int mrow  = lane & 15;

    v8f acc[2][4] = {};
    const bf16* arow0 = Ls + (size_t)(n0 + mrow) * NN;
    const bf16* arow1 = arow0 + (size_t)16 * NN;

    for (int kc = 0; kc < NN; kc += 32) {
        v16bf a0 = load_a_bf(arow0 + kc, khalf);
        v16bf a1 = load_a_bf(arow1 + kc, khalf);
        const bf16* brow = S + (size_t)kc * MM + j0 + (lane & 15);   // lane = N col
#pragma unroll
        for (int s = 0; s < 4; ++s) {
            v16bf bm = load_b_bf(brow + 16 * s, MM, khalf);
            acc[0][s] = WMMA_BF16(a0, bm, acc[0][s]);
            acc[1][s] = WMMA_BF16(a1, bm, acc[1][s]);
        }
    }
    // epilogue: C/D layout -> row = base + rr + 8*khalf, col = js + (lane&15)
    int nlo = lane & 15;
#pragma unroll
    for (int rt = 0; rt < 2; ++rt) {
#pragma unroll
        for (int s = 0; s < 4; ++s) {
            int col = j0 + 16 * s + nlo;
#pragma unroll
            for (int rr = 0; rr < 8; ++rr) {
                int row = n0 + 16 * rt + rr + 8 * khalf;
                float v = alpha * acc[rt][s][rr];
                if (t_mode) v += beta * bf2f(Tb[(size_t)row * MM + col]);
                Out[(size_t)row * MM + col] = f2bf(v);
            }
        }
    }
}

// ---------------- channel mix: out_gcn (+)= relu( [X|H1|H2|H3] @ W + b ) ----------------
__global__ __launch_bounds__(256) void mix_gemm(
    const bf16* __restrict__ Xb,
    const bf16* __restrict__ H1, const bf16* __restrict__ H2, const bf16* __restrict__ H3,
    const bf16* __restrict__ Wbf,        // [128][32]
    const float* __restrict__ bias, float* __restrict__ out_gcn, int accumulate)
{
    int wave = (blockIdx.x * blockDim.x + threadIdx.x) >> 5;
    int lane = threadIdx.x & 31;
    if (wave >= RR / 16) return;
    int r0 = wave * 16;
    int khalf = lane >> 4;
    int rA = r0 + (lane & 15);
    int b = rA / (NN * LL);
    int rem = rA - b * (NN * LL);
    int n = rem / LL;
    int t = rem - n * LL;
    size_t jrow = ((size_t)b * LL + t) * CC;   // column base inside [N][M] rows

    v8f acc[2] = {};
    const bf16* Hs[4] = {Xb, H1, H2, H3};
#pragma unroll
    for (int k = 0; k < 4; ++k) {
        const bf16* base = Hs[k] + (size_t)n * MM + jrow;
        v16bf a = load_a_bf(base, khalf);
#pragma unroll
        for (int s = 0; s < 2; ++s) {
            v16bf bm = load_b_bf(Wbf + ((size_t)(k * 32)) * 32 + 16 * s + (lane & 15), 32, khalf);
            acc[s] = WMMA_BF16(a, bm, acc[s]);
        }
    }
    int nlo = lane & 15;
#pragma unroll
    for (int s = 0; s < 2; ++s) {
        int o = 16 * s + nlo;
        float bia = bias[o];
#pragma unroll
        for (int rr = 0; rr < 8; ++rr) {
            int row = r0 + rr + 8 * khalf;
            float v = fmaxf(acc[s][rr] + bia, 0.f);
            size_t idx = (size_t)row * DD + o;
            if (accumulate) v += out_gcn[idx];
            out_gcn[idx] = v;
        }
    }
}

// ---------------- TCN causal 2x2 dilated conv as implicit GEMM (channels-last) ----------------
// out = relu(conv(src) + bias) [+ res]; dual-stores f32 (residual/gate use) and bf16 (next GEMM).
__global__ __launch_bounds__(256) void tcn_conv(
    const bf16* __restrict__ src, const bf16* __restrict__ Wt /*[128][32]*/,
    const float* __restrict__ bias, const float* __restrict__ res,
    float* __restrict__ out_f, bf16* __restrict__ out_bf, int dil)
{
    int wave = (blockIdx.x * blockDim.x + threadIdx.x) >> 5;
    int lane = threadIdx.x & 31;
    if (wave >= RR / 16) return;
    int r0 = wave * 16;
    int khalf = lane >> 4;
    int rA = r0 + (lane & 15);
    int b = rA / (NN * LL);
    int rem = rA - b * (NN * LL);
    int n = rem / LL;
    int l = rem - n * LL;
    bf16 zb = bfzero();

    v8f acc[2] = {};
#pragma unroll
    for (int tap = 0; tap < 4; ++tap) {
        int kn = tap >> 1, kl = tap & 1;
        int nn  = n - (1 - kn) * dil;
        int ll2 = l - (1 - kl) * dil;
        bool valid = (nn >= 0) && (ll2 >= 0);
        int nnc = valid ? nn : 0;
        int llc = valid ? ll2 : 0;
        const bf16* q = src + (((size_t)b * NN + nnc) * LL + llc) * CC;
        v8bf lo = *(const v8bf*)(q + khalf * 8);
        v8bf hi = *(const v8bf*)(q + 16 + khalf * 8);
        v16bf a;
#pragma unroll
        for (int e = 0; e < 8; ++e) {
            a[e]     = valid ? lo[e] : zb;
            a[8 + e] = valid ? hi[e] : zb;
        }
#pragma unroll
        for (int s = 0; s < 2; ++s) {
            v16bf bm = load_b_bf(Wt + ((size_t)(tap * 32)) * 32 + 16 * s + (lane & 15), 32, khalf);
            acc[s] = WMMA_BF16(a, bm, acc[s]);
        }
    }
    int nlo = lane & 15;
#pragma unroll
    for (int s = 0; s < 2; ++s) {
        int o = 16 * s + nlo;
        float bia = bias[o];
#pragma unroll
        for (int rr = 0; rr < 8; ++rr) {
            int row = r0 + rr + 8 * khalf;
            float v = fmaxf(acc[s][rr] + bia, 0.f);
            size_t idx = (size_t)row * DD + o;
            if (res) v += res[idx];
            if (out_f)  out_f[idx]  = v;
            if (out_bf) out_bf[idx] = f2bf(v);
        }
    }
}

// ---------------- gate fusion: out = relu(sig(fus@Wg+bg)*tcn + (1-sig)*gcn) ----------------
__global__ __launch_bounds__(256) void gate_fuse(
    const float* __restrict__ out_gcn, const float* __restrict__ out_tcn,
    const float* __restrict__ gw, const float* __restrict__ gb, float* __restrict__ out)
{
    int t = blockIdx.x * blockDim.x + threadIdx.x;   // RR*DD threads exactly
    int r = t >> 5, o = t & 31;
    const float* g1 = out_gcn + (size_t)r * 32;
    const float* t1 = out_tcn + (size_t)r * 32;
    float s = gb[o];
#pragma unroll 4
    for (int j = 0; j < 32; ++j) s += g1[j] * gw[j * 32 + o];
#pragma unroll 4
    for (int j = 0; j < 32; ++j) s += t1[j] * gw[(32 + j) * 32 + o];
    float gate = 1.f / (1.f + __expf(-s));
    float v = gate * t1[o] + (1.f - gate) * g1[o];
    out[t] = fmaxf(v, 0.f);
}

// ---------------- launch ----------------
extern "C" void kernel_launch(void* const* d_in, const int* in_sizes, int n_in,
                              void* d_out, int out_size, void* d_ws, size_t ws_size,
                              hipStream_t stream)
{
    const float* x_gcn  = (const float*)d_in[0];
    const float* x_tcn  = (const float*)d_in[1];
    const float* graphs = (const float*)d_in[2];
    const float* W_f    = (const float*)d_in[3];
    const float* b_f    = (const float*)d_in[4];
    const float* W_g    = (const float*)d_in[5];
    const float* b_g    = (const float*)d_in[6];
    const float* tcn_w  = (const float*)d_in[7];
    const float* tcn_b  = (const float*)d_in[8];
    const float* gate_w = (const float*)d_in[9];
    const float* gate_b = (const float*)d_in[10];
    float* out = (float*)d_out;
    (void)in_sizes; (void)n_in; (void)out_size; (void)ws_size;

    char* w = (char*)d_ws;
    auto take = [&](size_t n) -> char* { char* p = w; w += (n + 255) & ~(size_t)255; return p; };
    float* Lap     = (float*)take((size_t)NN * NN * 4);
    bf16*  Ls      = (bf16*) take((size_t)NN * NN * 2);
    bf16*  Xb      = (bf16*) take((size_t)NN * MM * 2);   // x_gcn packed [N][M]
    bf16*  H1      = (bf16*) take((size_t)NN * MM * 2);
    bf16*  H2      = (bf16*) take((size_t)NN * MM * 2);
    bf16*  H3      = (bf16*) take((size_t)NN * MM * 2);
    float* out_gcn = (float*)take((size_t)RR * DD * 4);
    bf16*  xt_bf   = (bf16*) take((size_t)RR * DD * 2);   // x_tcn in bf16 (identity layout)
    bf16*  a_bf    = (bf16*) take((size_t)RR * DD * 2);   // conv1/conv3 outputs (bf16 only)
    float* h1_f    = (float*)take((size_t)RR * DD * 4);   // level-0 output f32 (residual)
    bf16*  h1_bf   = (bf16*) take((size_t)RR * DD * 2);   // level-0 output bf16 (GEMM input)
    float* t_o     = (float*)take((size_t)RR * DD * 4);   // TCN final output f32
    float* dinv    = (float*)take(NN * 4);
    float* vv      = (float*)take(NN * 4);
    float* yy      = (float*)take(NN * 4);
    float* lmax    = (float*)take(256);
    bf16*  Wf_bf   = (bf16*) take(4096 * 2);
    bf16*  Wg_bf   = (bf16*) take(4096 * 2);
    bf16*  Wt      = (bf16*) take(4 * 4096 * 2);

    // weight + input conversions (W_f/W_g [k][c][o] is already GEMM-B layout)
    cvt_copy_bf<<<16, 256, 0, stream>>>(W_f, Wf_bf, 4096);
    cvt_copy_bf<<<16, 256, 0, stream>>>(W_g, Wg_bf, 4096);
    cvt_tcn_w<<<64, 256, 0, stream>>>(tcn_w, Wt);
    pack_x_bf<<<(NN * MM) / 256, 256, 0, stream>>>(x_gcn, Xb);
    cvt_copy_bf<<<(RR * DD) / 256, 256, 0, stream>>>(x_tcn, xt_bf, RR * DD);

    // ---- TCN: 2 levels, each conv1(relu) -> conv2(relu) -> +residual ----
    const int GT = (RR / 16) * 32 / 256;   // 3072 blocks
    tcn_conv<<<GT, 256, 0, stream>>>(xt_bf, Wt + 0 * 4096, tcn_b + 0,  nullptr, nullptr, a_bf, 1);
    tcn_conv<<<GT, 256, 0, stream>>>(a_bf,  Wt + 1 * 4096, tcn_b + 32, x_tcn,   h1_f, h1_bf, 1);
    tcn_conv<<<GT, 256, 0, stream>>>(h1_bf, Wt + 2 * 4096, tcn_b + 64, nullptr, nullptr, a_bf, 2);
    tcn_conv<<<GT, 256, 0, stream>>>(a_bf,  Wt + 3 * 4096, tcn_b + 96, h1_f,    t_o, nullptr, 2);

    // ---- ChebNet per graph ----
    const int GP = (NN / 32) * (MM / 64) * 32 / 256;  // 768 blocks
    for (int g = 0; g < 2; ++g) {
        const float* A = graphs + (size_t)g * NN * NN;
        deg_kernel<<<256, 256, 0, stream>>>(A, dinv);
        build_lap<<<NN * NN / 256, 256, 0, stream>>>(A, dinv, Lap);
        init_v<<<8, 256, 0, stream>>>(vv);
        for (int it = 0; it < 25; ++it) {            // power iteration -> lambda_max
            matvec<<<256, 256, 0, stream>>>(Lap, vv, yy);
            norm_scale<<<1, 256, 0, stream>>>(yy, vv, lmax);
        }
        build_ls<<<NN * NN / 256, 256, 0, stream>>>(Lap, lmax, Ls);

        // Chebyshev recurrence: H1 = Ls@X ; H2 = 2*Ls@H1 - X ; H3 = 2*Ls@H2 - H1
        prop_gemm<<<GP, 256, 0, stream>>>(Ls, Xb, nullptr, 0, 1.f,  0.f, H1);
        prop_gemm<<<GP, 256, 0, stream>>>(Ls, H1, Xb,      1, 2.f, -1.f, H2);
        prop_gemm<<<GP, 256, 0, stream>>>(Ls, H2, H1,      1, 2.f, -1.f, H3);

        // out_gcn (+)= relu([X|H1|H2|H3] @ W + b)
        mix_gemm<<<GT, 256, 0, stream>>>(Xb, H1, H2, H3,
                                         g ? Wg_bf : Wf_bf, g ? b_g : b_f, out_gcn, g);
    }

    // ---- gated fusion ----
    gate_fuse<<<(RR * DD) / 256, 256, 0, stream>>>(out_gcn, t_o, gate_w, gate_b, out);
}
